// DNN1_59966333387212
// MI455X (gfx1250) — compile-verified, matrix operand order checked by reading, not run on hardware
//
#include <hip/hip_runtime.h>
#include <hip/hip_bf16.h>

typedef __bf16 bf16_t;
typedef __attribute__((ext_vector_type(16))) __bf16 v16bf;
typedef __attribute__((ext_vector_type(8)))  __bf16 v8bf;
typedef __attribute__((ext_vector_type(8)))  float  v8f;

#define BK   32
#define LDT  40   // BK + 8 pad; 2*LDT % 16 == 0 so rows stay 16B aligned

// ---------------------------------------------------------------------------
// CDNA5 async global->LDS 128-bit copy (ISA §10: GLOBAL_LOAD_ASYNC_TO_LDS_B128,
// VDST = LDS byte address VGPR, VADDR = 64-bit global address, ASYNCcnt).
// Generic pointers to LDS carry the LDS offset in their low 32 bits.
// ---------------------------------------------------------------------------
__device__ __forceinline__ void async_ld_b128(void* lds, const void* g) {
    asm volatile("global_load_async_to_lds_b128 %0, %1, off"
                 :: "v"((unsigned)(unsigned long long)lds), "v"(g)
                 : "memory");
}
__device__ __forceinline__ void async_wait0() {
    asm volatile("s_wait_asynccnt 0x0" ::: "memory");
}

// ---------------------------------------------------------------------------
// Fragment loader: 16x32 bf16 tile from LDS, rows [row][LDT] K-contiguous.
// CDNA5 16-bit A/B layout: lane L (L%16 = row/col), elements 0..7 are
// K = (L<16?0:8)+0..7, elements 8..15 are K = 16+(L<16?0:8)+0..7.
// Two ds_load_b128 per fragment.
// ---------------------------------------------------------------------------
__device__ __forceinline__ v16bf load_frag(const bf16_t* sm, int row0, int lane) {
    int r  = row0 + (lane & 15);
    int kb = (lane >> 4) << 3;               // 0 or 8
    const v8bf* p0 = (const v8bf*)(sm + r * LDT + kb);
    const v8bf* p1 = (const v8bf*)(sm + r * LDT + kb + 16);
    v16bf o;
#pragma unroll
    for (int i = 0; i < 8; ++i) { o[i] = (*p0)[i]; o[i + 8] = (*p1)[i]; }
    return o;
}

// ---------------------------------------------------------------------------
// Batched NT-GEMM: C[b] = relu(A[b](MxK) * Bt[b](NxK)^T + bias[m]).
// A, Bt bf16 row-major with K contiguous -> all staging is 128-bit.
// transpose_out=0: C[m][n] row-major (ldc = N-stride).
// transpose_out=1: C[n][m] row-major (ldc = M-stride), vectorized v8bf stores.
// Block tile 128x128, K-chunk 32, 8 waves (4M x 2N), wave tile 32x64.
// ---------------------------------------------------------------------------
__global__ __launch_bounds__(256) void gemm_nt_bias_relu(
    const bf16_t* __restrict__ A,  int lda, long long strideA,
    const bf16_t* __restrict__ Bt, int ldb, long long strideB,
    bf16_t*       __restrict__ C,  int ldc, long long strideC, int transpose_out,
    const float*  __restrict__ bias, int K)
{
    __shared__ __align__(16) bf16_t Asm[128 * LDT];
    __shared__ __align__(16) bf16_t Bsm[128 * LDT];   // [n][k]

    int b = blockIdx.z;
    const bf16_t* Ab  = A  + (size_t)b * strideA;
    const bf16_t* Btb = Bt + (size_t)b * strideB;
    bf16_t*       Cb  = C  + (size_t)b * strideC;

    int m0 = blockIdx.x * 128, n0 = blockIdx.y * 128;
    int tid = threadIdx.x, lane = tid & 31, wid = tid >> 5;
    int wm = (wid >> 1) * 32;       // 0,32,64,96
    int wn = (wid & 1) * 64;        // 0,64

    v8f acc[2][4];
#pragma unroll
    for (int i = 0; i < 2; ++i)
#pragma unroll
        for (int j = 0; j < 4; ++j) acc[i][j] = (v8f)(0.f);

    for (int k0 = 0; k0 < K; k0 += BK) {
        // stage both tiles with 16B vectors: 128 rows x 4 vec/row, 2 per thread
#pragma unroll
        for (int i = 0; i < 2; ++i) {
            int v = tid + i * 256;
            int r = v >> 2, kv = (v & 3) << 3;
            *(v8bf*)&Asm[r * LDT + kv] =
                *(const v8bf*)(Ab  + (size_t)(m0 + r) * lda + k0 + kv);
            *(v8bf*)&Bsm[r * LDT + kv] =
                *(const v8bf*)(Btb + (size_t)(n0 + r) * ldb + k0 + kv);
        }
        __syncthreads();
#pragma unroll
        for (int mt = 0; mt < 2; ++mt) {
            v16bf a = load_frag(Asm, wm + mt * 16, lane);
#pragma unroll
            for (int nt = 0; nt < 4; ++nt) {
                v16bf bb = load_frag(Bsm, wn + nt * 16, lane);
                acc[mt][nt] = __builtin_amdgcn_wmma_f32_16x16x32_bf16(
                    false, a, false, bb, (short)0, acc[mt][nt], false, false);
            }
        }
        __syncthreads();
    }

    // epilogue: C-frag layout lane<16 -> M=r, lane>=16 -> M=r+8; N = lane&15
    int mhi = (lane >> 4) << 3;
    if (!transpose_out) {
#pragma unroll
        for (int mt = 0; mt < 2; ++mt)
#pragma unroll
            for (int nt = 0; nt < 4; ++nt)
#pragma unroll
                for (int r = 0; r < 8; ++r) {
                    int m = m0 + wm + mt * 16 + r + mhi;
                    int n = n0 + wn + nt * 16 + (lane & 15);
                    float v = acc[mt][nt][r] + bias[m];
                    Cb[(size_t)m * ldc + n] = (bf16_t)(v > 0.f ? v : 0.f);
                }
    } else {
#pragma unroll
        for (int mt = 0; mt < 2; ++mt) {
            int mbase = m0 + wm + mt * 16 + mhi;     // multiple of 8
#pragma unroll
            for (int nt = 0; nt < 4; ++nt) {
                int n = n0 + wn + nt * 16 + (lane & 15);
                v8bf pack;
#pragma unroll
                for (int r = 0; r < 8; ++r) {
                    float v = acc[mt][nt][r] + bias[mbase + r];
                    pack[r] = (bf16_t)(v > 0.f ? v : 0.f);
                }
                *(v8bf*)(Cb + (size_t)n * ldc + mbase) = pack;
            }
        }
    }
}

// ---------------------------------------------------------------------------
// Fused LSTM step: g = [seq_t ; h_in](128x512) x wcat(1024x512)^T + bias.
// Grid 8: block owns 32 hidden units j across all 4 gates; each wave holds
// i,f,g,o of its (b,j) block in registers -> in-register cell update.
// Staging uses CDNA5 async global->LDS b128 (no VGPR round-trip).
// h double-buffered across steps.
// ---------------------------------------------------------------------------
__global__ __launch_bounds__(256) void lstm_step(
    const bf16_t* __restrict__ seq,   // [B][131072]; row t of b at b*131072+t*256
    int t,
    const bf16_t* __restrict__ wcat,  // [1024][512]: row n = {w_ih[n][:], w_hh[n][:]}
    const float*  __restrict__ bias,  // [1024] = b_ih + b_hh
    const bf16_t* __restrict__ h_in,  // [128][256]
    bf16_t*       __restrict__ h_out, // [128][256]
    float*        __restrict__ c,     // [128][256] (exclusive per-element RMW)
    float*        __restrict__ out,   // nullptr, or [128][512]
    int out_off)
{
    __shared__ __align__(16) bf16_t Asm[128 * LDT];
    __shared__ __align__(16) bf16_t Bsm[128 * LDT];   // [c][k], c = gate*32 + (j-j0)

    int tid = threadIdx.x, lane = tid & 31, wid = tid >> 5;
    int j0 = blockIdx.x * 32;
    int wm = (wid >> 1) * 32;        // batch group: 0,32,64,96
    int jw = (wid & 1) * 16;         // j sub-group: 0,16

    v8f acc[2][4];
#pragma unroll
    for (int i = 0; i < 2; ++i)
#pragma unroll
        for (int g = 0; g < 4; ++g) acc[i][g] = (v8f)(0.f);

    for (int k0 = 0; k0 < 512; k0 += BK) {
#pragma unroll
        for (int i = 0; i < 2; ++i) {
            int v = tid + i * 256;
            int r = v >> 2, kv = (v & 3) << 3;
            int k = k0 + kv;                     // 8-chunk never crosses 256
            // A: rows = batch, K = concat(seq_t, h_in)
            const bf16_t* srcA = (k < 256)
                ? seq  + (size_t)r * 131072 + (size_t)t * 256 + k
                : h_in + r * 256 + (k - 256);
            async_ld_b128(&Asm[r * LDT + kv], srcA);
            // B: row c -> gate*256 + j0 + (c&31) of wcat
            int n = ((r >> 5) << 8) + j0 + (r & 31);
            async_ld_b128(&Bsm[r * LDT + kv], wcat + (size_t)n * 512 + k);
        }
        async_wait0();
        __syncthreads();
#pragma unroll
        for (int mt = 0; mt < 2; ++mt) {
            v16bf a = load_frag(Asm, wm + mt * 16, lane);
#pragma unroll
            for (int g = 0; g < 4; ++g) {
                v16bf bb = load_frag(Bsm, g * 32 + jw, lane);
                acc[mt][g] = __builtin_amdgcn_wmma_f32_16x16x32_bf16(
                    false, a, false, bb, (short)0, acc[mt][g], false, false);
            }
        }
        __syncthreads();
    }

    // cell update: i,f,g,o for (b, j) all in this wave's registers
    int j   = j0 + jw + (lane & 15);
    int mhi = (lane >> 4) << 3;
    float bi = bias[0 * 256 + j], bf = bias[1 * 256 + j];
    float bg = bias[2 * 256 + j], bo = bias[3 * 256 + j];
#pragma unroll
    for (int mt = 0; mt < 2; ++mt)
#pragma unroll
        for (int r = 0; r < 8; ++r) {
            int b = wm + mt * 16 + r + mhi;
            float gi = acc[mt][0][r] + bi;
            float gf = acc[mt][1][r] + bf;
            float gg = acc[mt][2][r] + bg;
            float go = acc[mt][3][r] + bo;
            float si = 1.f / (1.f + __expf(-gi));
            float sf = 1.f / (1.f + __expf(-gf));
            float so = 1.f / (1.f + __expf(-go));
            float cn = sf * c[b * 256 + j] + si * tanhf(gg);
            c[b * 256 + j] = cn;
            float hn = so * tanhf(cn);
            h_out[b * 256 + j] = (bf16_t)hn;
            if (out) out[(size_t)b * 512 + out_off + j] = hn;
        }
}

// ------------------------------ prep kernels -------------------------------
__global__ void k_f32_to_bf16(const float* __restrict__ s, bf16_t* __restrict__ d, int n) {
    int i = blockIdx.x * blockDim.x + threadIdx.x;
    if (i < n) d[i] = (bf16_t)s[i];
}
// wcat[n][k] (1024x512): k<256 -> w_ih[n][k] ; k>=256 -> w_hh[n][k-256]
__global__ void k_build_wcat(const float* __restrict__ wih, const float* __restrict__ whh,
                             bf16_t* __restrict__ d) {
    int i = blockIdx.x * blockDim.x + threadIdx.x;
    if (i >= 1024 * 512) return;
    int n = i >> 9, k = i & 511;
    float v = (k < 256) ? wih[n * 256 + k] : whh[n * 256 + (k - 256)];
    d[i] = (bf16_t)v;
}
__global__ void k_add2(const float* __restrict__ a, const float* __restrict__ b,
                       float* __restrict__ d, int n) {
    int i = blockIdx.x * blockDim.x + threadIdx.x;
    if (i < n) d[i] = a[i] + b[i];
}
__global__ void k_zero32(unsigned* __restrict__ p, int nwords) {
    int i = blockIdx.x * blockDim.x + threadIdx.x;
    if (i < nwords) p[i] = 0u;
}
// x[b][c][t] f32 -> x_t[b][t][c] bf16 via 32x32 LDS tiles
__global__ __launch_bounds__(256) void k_transpose_x(const float* __restrict__ x,
                                                     bf16_t* __restrict__ xt) {
    __shared__ float tile[32][33];
    int b = blockIdx.z;
    int c0 = blockIdx.x * 32, t0 = blockIdx.y * 32;
    int tx = threadIdx.x & 31, ty = threadIdx.x >> 5;   // 32 x 8
    const float* xb  = x  + (size_t)b * 262144;
    bf16_t*      xtb = xt + (size_t)b * 262144;
#pragma unroll
    for (int i = 0; i < 32; i += 8)
        tile[ty + i][tx] = xb[(size_t)(c0 + ty + i) * 512 + t0 + tx];
    __syncthreads();
#pragma unroll
    for (int i = 0; i < 32; i += 8)
        xtb[(size_t)(t0 + ty + i) * 512 + c0 + tx] = (bf16_t)tile[tx][ty + i];
}

// ---------------------------------------------------------------------------
extern "C" void kernel_launch(void* const* d_in, const int* in_sizes, int n_in,
                              void* d_out, int out_size, void* d_ws, size_t ws_size,
                              hipStream_t stream) {
    (void)in_sizes; (void)n_in; (void)out_size; (void)ws_size;
    const float* x      = (const float*)d_in[0];   // [128,512,512,1]
    const float* w1     = (const float*)d_in[1];   // [256,512]
    const float* b1     = (const float*)d_in[2];
    const float* w2     = (const float*)d_in[3];   // [256,256]
    const float* b2     = (const float*)d_in[4];
    const float* w_ih_f = (const float*)d_in[5];   // [1024,256]
    const float* w_hh_f = (const float*)d_in[6];
    const float* b_ih_f = (const float*)d_in[7];
    const float* b_hh_f = (const float*)d_in[8];
    const float* w_ih_b = (const float*)d_in[9];
    const float* w_hh_b = (const float*)d_in[10];
    const float* b_ih_b = (const float*)d_in[11];
    const float* b_hh_b = (const float*)d_in[12];
    float* out = (float*)d_out;                    // [128, 512]

    // workspace layout (bytes)
    char* ws = (char*)d_ws;
    bf16_t* x_t    = (bf16_t*)(ws);                 // [B][T][CIN]   67.11MB
    bf16_t* y1_t   = (bf16_t*)(ws +  67108864);     // [B][T][H]     33.55MB
    bf16_t* y2     = (bf16_t*)(ws + 100663296);     // [B][H][T]     33.55MB (seq view)
    bf16_t* wcat_f = (bf16_t*)(ws + 134217728);     // [1024][512]   1MB
    bf16_t* wcat_b = (bf16_t*)(ws + 135266304);     // 1MB
    bf16_t* w1b    = (bf16_t*)(ws + 136314880);     // 256KB
    bf16_t* w2b    = (bf16_t*)(ws + 136577024);     // 128KB
    float*  bias_f = (float*) (ws + 136708096);     // 4KB
    float*  bias_b = (float*) (ws + 136712192);     // 4KB
    bf16_t* h_f0   = (bf16_t*)(ws + 136716288);     // 64KB
    bf16_t* h_f1   = (bf16_t*)(ws + 136781824);     // 64KB
    float*  c_f    = (float*) (ws + 136847360);     // 128KB
    bf16_t* h_b0   = (bf16_t*)(ws + 136978432);     // 64KB
    bf16_t* h_b1   = (bf16_t*)(ws + 137043968);     // 64KB
    float*  c_b    = (float*) (ws + 137109504);     // 128KB

    // ---- prep ----
    k_transpose_x<<<dim3(16, 16, 128), 256, 0, stream>>>(x, x_t);
    k_f32_to_bf16<<<(256 * 512 + 255) / 256, 256, 0, stream>>>(w1, w1b, 256 * 512);
    k_f32_to_bf16<<<(256 * 256 + 255) / 256, 256, 0, stream>>>(w2, w2b, 256 * 256);
    k_build_wcat<<<(1024 * 512 + 255) / 256, 256, 0, stream>>>(w_ih_f, w_hh_f, wcat_f);
    k_build_wcat<<<(1024 * 512 + 255) / 256, 256, 0, stream>>>(w_ih_b, w_hh_b, wcat_b);
    k_add2<<<4, 256, 0, stream>>>(b_ih_f, b_hh_f, bias_f, 1024);
    k_add2<<<4, 256, 0, stream>>>(b_ih_b, b_hh_b, bias_b, 1024);
    k_zero32<<<64, 256, 0, stream>>>((unsigned*)h_f0, 16384);
    k_zero32<<<128, 256, 0, stream>>>((unsigned*)c_f, 32768);
    k_zero32<<<64, 256, 0, stream>>>((unsigned*)h_b0, 16384);
    k_zero32<<<128, 256, 0, stream>>>((unsigned*)c_b, 32768);

    // ---- conv1: y1_t[b][t][h] = relu(w1b(256x512) . x_t[b](512x512)^T + b1)
    gemm_nt_bias_relu<<<dim3(2, 4, 128), 256, 0, stream>>>(
        w1b, 512, 0, x_t, 512, 262144, y1_t, 256, 131072, 1, b1, 512);

    // ---- conv2: y2[b][h][t] = relu(w2b(256x256) . y1_t[b](512x256)^T + b2)
    gemm_nt_bias_relu<<<dim3(2, 4, 128), 256, 0, stream>>>(
        w2b, 256, 0, y1_t, 256, 131072, y2, 512, 131072, 0, b2, 256);

    // ---- forward LSTM: 512 sequential fused steps (h double-buffered) ----
    for (int t = 0; t < 512; ++t) {
        bf16_t* hin  = (t & 1) ? h_f1 : h_f0;
        bf16_t* hout = (t & 1) ? h_f0 : h_f1;
        lstm_step<<<8, 256, 0, stream>>>(y2, t, wcat_f, bias_f, hin, hout, c_f,
                                         (t == 511) ? out : (float*)nullptr, 0);
    }

    // ---- backward direction: out[:, -1, H:] == one LSTM step from zero state
    //      on seq[:, T-1, :]  (h_b0 and c_b are zero)
    lstm_step<<<8, 256, 0, stream>>>(y2, 511, wcat_b, bias_b, h_b0, h_b1, c_b,
                                     out, 256);
}
